// TCMHGT_34600256537271
// MI455X (gfx1250) — matrix operand order repaired and checked by reading, not compiled, not run on hardware
//
#include <hip/hip_runtime.h>
#include <cmath>

// ---------------------------------------------------------------------------
// Problem constants (match the reference)
// ---------------------------------------------------------------------------
#define T_   2
#define N_   10000
#define E_   60000
#define H_   8
#define FIN_ 2048
#define HID_ 512
#define OUT_ 8

static const int EDGE_ST[4] = {0, 0, 1, 1};
static const int EDGE_DT[4] = {0, 1, 0, 1};

// ---------------------------------------------------------------------------
// Types for WMMA / TDM
// ---------------------------------------------------------------------------
typedef __attribute__((ext_vector_type(16))) __bf16 v16bf;
typedef __attribute__((ext_vector_type(8)))  float  v8f;

typedef __attribute__((ext_vector_type(4))) unsigned int tdm_v4u;
typedef __attribute__((ext_vector_type(8))) int          tdm_v8i;
typedef __attribute__((ext_vector_type(4))) int          tdm_v4i;

#if defined(__has_builtin)
#  if __has_builtin(__builtin_amdgcn_tensor_load_to_lds) && \
      __has_builtin(__builtin_amdgcn_s_wait_tensorcnt)
#    define HAVE_TDM 1
#  endif
#endif

union FragU {
  unsigned int u[8];
  v16bf v;
};

__device__ __forceinline__ unsigned short f2bf(float f) {
  // round-to-nearest-even f32 -> bf16
  unsigned int u = __float_as_uint(f);
  u += 0x7FFFu + ((u >> 16) & 1u);
  return (unsigned short)(u >> 16);
}

__device__ __forceinline__ void atomicMaxFloat(float* addr, float v) {
  // int-max for non-negative, uint-min for negative: both monotone wrt float order.
  if (v >= 0.0f) {
    atomicMax((int*)addr, __float_as_int(v));
  } else {
    atomicMin((unsigned int*)addr, __float_as_uint(v));
  }
}

// ---------------------------------------------------------------------------
// Generic batched GEMM: C[z] = A[z] (MxK, f32) * W[z] (KxNout, f32) + bias[z]
// bf16 WMMA 16x16x32, f32 accumulate.  Block = 256 threads (8 wave32 waves),
// tile 128(M) x 64(N), K-step 32.  Nout % 64 == 0, K % 32 == 0 required.
// B tile is fetched with the Tensor Data Mover (tensor_load_to_lds) when the
// builtin is available, then converted/transposed f32->bf16 in LDS.
// ---------------------------------------------------------------------------
__global__ void __launch_bounds__(256)
TCMHGT_gemm_bf16(const float* __restrict__ A, long long lda, long long sA,
                 const float* __restrict__ W, long long ldw, long long sW,
                 const float* __restrict__ bias, long long sBias,
                 float* __restrict__ C, long long ldc, long long sC,
                 int M, int Nout, int K)
{
  __shared__ unsigned short Abuf[128 * 40];  // 128 rows (M) x 32 (K) bf16, pitch 40
  __shared__ unsigned short Bbuf[64 * 40];   // 64 cols (N) x 32 (K) bf16, transposed
#if HAVE_TDM
  __shared__ float Braw[32 * 64];            // raw f32 B tile, TDM destination
#endif

  const int tid   = threadIdx.x;
  const int lane  = tid & 31;
  const int wave  = tid >> 5;
  const int lh    = (lane >> 4) & 1;         // lane half (wave32)
  const int ln    = lane & 15;
  const int tileM = blockIdx.x * 128;
  const int tileN = blockIdx.y * 64;
  const long long z = blockIdx.z;

  A += z * sA;
  W += z * sW;
  C += z * sC;
  if (bias) bias += z * sBias;

  v8f acc[4] = {};

  const int nk = K >> 5;
  for (int kb = 0; kb < nk; ++kb) {
    const int k0 = kb << 5;

#if HAVE_TDM
    // ---- TDM: async DMA of B tile W[k0:k0+32, tileN:tileN+64] f32 -> Braw.
    // Descriptor per CDNA5 ISA sec. 8.3/8.4; issued once per block (wave 0;
    // condition is wave-uniform so this is a single tensor_load_to_lds).
    if (wave == 0) {
      const float* gsrc = W + (long long)k0 * ldw + tileN;
      unsigned long long ga = (unsigned long long)(size_t)gsrc;
      unsigned lds_off = (unsigned)(size_t)(void*)&Braw[0]; // low 32 bits = LDS byte offset
      unsigned dim0 = (unsigned)(Nout - tileN);  // remaining extent for OOB zero-fill
      unsigned dim1 = (unsigned)(K - k0);
      tdm_v4u g0;
      g0[0] = 1u;                                          // count=1, user mode
      g0[1] = lds_off;                                     // lds_addr
      g0[2] = (unsigned)(ga & 0xFFFFFFFFu);                // global_addr[31:0]
      g0[3] = (unsigned)((ga >> 32) & 0x01FFFFFFu) |       // global_addr[56:32]
              (2u << 30);                                  // type=2 (image)
      tdm_v8i g1;
      g1[0] = (int)(2u << 16);                             // wg_mask=0, data_size=4B
      g1[1] = (int)((dim0 & 0xFFFFu) << 16);               // tensor_dim0[15:0]
      g1[2] = (int)(((dim0 >> 16) & 0xFFFFu) |             // tensor_dim0[31:16]
                    ((dim1 & 0xFFFFu) << 16));             // tensor_dim1[15:0]
      g1[3] = (int)(((dim1 >> 16) & 0xFFFFu) |             // tensor_dim1[31:16]
                    (64u << 16));                          // tile_dim0 = 64
      g1[4] = 32;                                          // tile_dim1 = 32, tile_dim2 = 0
      g1[5] = (int)(unsigned)ldw;                          // tensor_dim0_stride[31:0]
      g1[6] = 0;                                           // stride hi / dim1_stride lo
      g1[7] = 0;
      tdm_v4i gz = {0, 0, 0, 0};
#if defined(__clang_major__) && __clang_major__ >= 23
      tdm_v8i gz8 = {0, 0, 0, 0, 0, 0, 0, 0};
      __builtin_amdgcn_tensor_load_to_lds(g0, g1, gz, gz, gz8, 0);
#else
      __builtin_amdgcn_tensor_load_to_lds(g0, g1, gz, gz, 0);
#endif
    }
#endif  // HAVE_TDM

    // ---- stage A (128x32 f32 -> bf16) with 16B vector loads ---------------
#pragma unroll
    for (int i = 0; i < 4; ++i) {
      int idx = tid + (i << 8);            // 0..1023
      int r   = idx >> 3;                  // 0..127
      int c4  = (idx & 7) << 2;            // 0,4,...,28
      int gr  = tileM + r;
      float4 av = (gr < M) ? *(const float4*)&A[(long long)gr * lda + (k0 + c4)]
                           : make_float4(0.0f, 0.0f, 0.0f, 0.0f);
      uint2 pk;
      pk.x = (unsigned)f2bf(av.x) | ((unsigned)f2bf(av.y) << 16);
      pk.y = (unsigned)f2bf(av.z) | ((unsigned)f2bf(av.w) << 16);
      *(uint2*)&Abuf[r * 40 + c4] = pk;
    }
    if (k0 + 32 < K) {                     // speculative prefetch of next A tile
      int pr = tileM + tid;
      if (pr >= M) pr = M - 1;
      __builtin_prefetch(&A[(long long)pr * lda + (k0 + 32)], 0, 1);
    }

#if HAVE_TDM
    if (wave == 0) __builtin_amdgcn_s_wait_tensorcnt(0);
    __syncthreads();                       // Braw ready for all waves
    // ---- convert/transpose B: Braw[k][n] f32 -> Bbuf[n][k] bf16 -----------
#pragma unroll
    for (int i = 0; i < 4; ++i) {
      int idx = tid + (i << 8);            // 0..1023
      int nn  = idx & 63;
      int kk2 = (idx >> 6) << 1;           // 0,2,...,30
      unsigned p = (unsigned)f2bf(Braw[kk2 * 64 + nn]) |
                   ((unsigned)f2bf(Braw[(kk2 + 1) * 64 + nn]) << 16);
      *(unsigned*)&Bbuf[nn * 40 + kk2] = p;
    }
#else
    // ---- fallback: direct global staging of B (transposed) ----------------
#pragma unroll
    for (int i = 0; i < 8; ++i) {
      int idx = tid + (i << 8);            // 0..2047
      int nn  = idx >> 5;                  // 0..63 output column
      int cc  = idx & 31;                  // k within tile
      Bbuf[nn * 40 + cc] = f2bf(W[(long long)(k0 + cc) * ldw + (tileN + nn)]);
    }
#endif
    __syncthreads();

    // ---- A fragment: 16x32 bf16, ISA 7.12.2 layout ------------------------
    FragU afrag;
    {
      const unsigned short* Ar = &Abuf[(wave * 16 + ln) * 40];
#pragma unroll
      for (int vr = 0; vr < 8; ++vr) {
        int kk = (vr < 4 ? (vr << 1) : (16 + ((vr - 4) << 1))) + (lh << 3);
        afrag.u[vr] = *(const unsigned int*)&Ar[kk];
      }
    }
    // ---- preload all 4 B fragments, then 4 back-to-back WMMAs -------------
    FragU bfrag[4];
#pragma unroll
    for (int f = 0; f < 4; ++f) {
      const unsigned short* Bc = &Bbuf[(f * 16 + ln) * 40];
#pragma unroll
      for (int vr = 0; vr < 8; ++vr) {
        int kk = (vr < 4 ? (vr << 1) : (16 + ((vr - 4) << 1))) + (lh << 3);
        bfrag[f].u[vr] = *(const unsigned int*)&Bc[kk];
      }
    }
#pragma unroll
    for (int f = 0; f < 4; ++f) {
      acc[f] = __builtin_amdgcn_wmma_f32_16x16x32_bf16(
          false, afrag.v, false, bfrag[f].v, (short)0, acc[f], false, false);
    }
    __syncthreads();
  }

  // --- store: D layout lanes 0-15 -> M=r, lanes 16-31 -> M=r+8 -------------
  const int rbase = tileM + wave * 16 + lh * 8;
#pragma unroll
  for (int f = 0; f < 4; ++f) {
    const int gc = tileN + f * 16 + ln;
    const float bv = bias ? bias[gc] : 0.0f;
#pragma unroll
    for (int r = 0; r < 8; ++r) {
      int gr = rbase + r;
      if (gr < M) C[(long long)gr * ldc + gc] = acc[f][r] + bv;
    }
  }
}

// ---------------------------------------------------------------------------
// Utility / edge kernels
// ---------------------------------------------------------------------------
__global__ void TCMHGT_fill_u32(unsigned int* p, unsigned int v, size_t n) {
  size_t i = (size_t)blockIdx.x * blockDim.x + threadIdx.x;
  size_t st = (size_t)gridDim.x * blockDim.x;
  for (; i < n; i += st) p[i] = v;
}

__global__ void TCMHGT_edge_logits(const float* __restrict__ q,
                                   const float* __restrict__ krel,
                                   const int* __restrict__ src,
                                   const int* __restrict__ dst,
                                   const float* __restrict__ prel,
                                   float* __restrict__ alpha,
                                   float* __restrict__ segmax,
                                   int E, int H, int D, int ld, float inv_sqrt_d)
{
  int idx = blockIdx.x * blockDim.x + threadIdx.x;
  if (idx >= E * H) return;
  int i = idx / H, h = idx - i * H;
  int s = src[i], d = dst[i];
  const float* qp = q + (long long)d * ld + h * D;
  const float* kp = krel + (long long)s * ld + h * D;
  float acc = 0.0f;
  for (int j = 0; j < D; ++j) acc = fmaf(qp[j], kp[j], acc);
  acc *= prel[h] * inv_sqrt_d;
  alpha[idx] = acc;
  atomicMaxFloat(&segmax[(long long)d * H + h], acc);
}

__global__ void TCMHGT_edge_exp(float* __restrict__ alpha,
                                const float* __restrict__ segmax,
                                float* __restrict__ segsum,
                                const int* __restrict__ dst, int E, int H)
{
  int idx = blockIdx.x * blockDim.x + threadIdx.x;
  if (idx >= E * H) return;
  int i = idx / H, h = idx - i * H;
  int d = dst[i];
  float v = expf(alpha[idx] - segmax[(long long)d * H + h]);
  alpha[idx] = v;
  atomicAdd(&segsum[(long long)d * H + h], v);
}

__global__ void TCMHGT_edge_agg(const float* __restrict__ alpha,
                                const float* __restrict__ segsum,
                                const float* __restrict__ vrel,
                                const int* __restrict__ src,
                                const int* __restrict__ dst,
                                float* __restrict__ agg,
                                int H, int D, int ld)
{
  int i = blockIdx.x;                  // one block per edge
  int s = src[i], d = dst[i];
  int HD = H * D;
  for (int j = threadIdx.x; j < HD; j += blockDim.x) {
    int h = j / D;
    float w = alpha[(long long)i * H + h] /
              (segsum[(long long)d * H + h] + 1e-16f);
    atomicAdd(&agg[(long long)d * ld + j], w * vrel[(long long)s * ld + j]);
  }
}

__global__ void TCMHGT_gelu(const float* __restrict__ x, float* __restrict__ y, size_t n) {
  size_t i = (size_t)blockIdx.x * blockDim.x + threadIdx.x;
  size_t st = (size_t)gridDim.x * blockDim.x;
  for (; i < n; i += st) {
    float v = x[i];
    y[i] = 0.5f * v * (1.0f + erff(v * 0.70710678118654752f));
  }
}

__global__ void TCMHGT_head(const float* __restrict__ h2,
                            const float* __restrict__ Wlin,
                            const float* __restrict__ blin,
                            float* __restrict__ out, int NT, int F)
{
  int n = blockIdx.x * blockDim.x + threadIdx.x;
  if (n >= NT) return;
  const float* row = h2 + (long long)n * F;
  float lg[OUT_];
#pragma unroll
  for (int o = 0; o < OUT_; ++o) lg[o] = blin[o];
  for (int k = 0; k < F; ++k) {
    float xv = row[k];
#pragma unroll
    for (int o = 0; o < OUT_; ++o) lg[o] = fmaf(xv, Wlin[k * OUT_ + o], lg[o]);
  }
  float m = lg[0];
#pragma unroll
  for (int o = 1; o < OUT_; ++o) m = fmaxf(m, lg[o]);
  float s = 0.0f;
#pragma unroll
  for (int o = 0; o < OUT_; ++o) { float e = expf(lg[o] - m); lg[o] = e; s += e; }
  float inv = 1.0f / s;
#pragma unroll
  for (int o = 0; o < OUT_; ++o) out[(long long)n * OUT_ + o] = lg[o] * inv;
}

// ---------------------------------------------------------------------------
// Host helpers
// ---------------------------------------------------------------------------
static void launch_gemm(hipStream_t s,
                        const float* A, long long lda, long long sA,
                        const float* W, long long ldw, long long sW,
                        const float* bias, long long sBias,
                        float* C, long long ldc, long long sC,
                        int M, int Nout, int K, int batch)
{
  dim3 g((unsigned)((M + 127) / 128), (unsigned)(Nout / 64), (unsigned)batch);
  TCMHGT_gemm_bf16<<<g, 256, 0, s>>>(A, lda, sA, W, ldw, sW, bias, sBias,
                                     C, ldc, sC, M, Nout, K);
}

static void fillu(hipStream_t s, void* p, unsigned v, size_t n) {
  size_t blocks = (n + 255) / 256;
  if (blocks > 4096) blocks = 4096;
  if (blocks == 0) blocks = 1;
  TCMHGT_fill_u32<<<(unsigned)blocks, 256, 0, s>>>((unsigned int*)p, v, n);
}

static void run_layer(hipStream_t s,
                      const float* xin, int Fin, int Fout, int D,
                      const float* Wk, const float* bk,
                      const float* Wq, const float* bq,
                      const float* Wv, const float* bv,
                      const float* Wa, const float* ba,
                      const float* arel, const float* mrel, const float* prel,
                      const int* const* edges,
                      float* kbuf, float* qbuf, float* vbuf, float* aggbuf,
                      float* gbuf, float* hbuf,
                      float* krel, float* vrel,
                      float* alpha, float* segmax, float* segsum)
{
  // K/Q/V projections, batched over node types
  launch_gemm(s, xin, Fin, (long long)N_ * Fin, Wk, Fout, (long long)Fin * Fout,
              bk, Fout, kbuf, Fout, (long long)N_ * Fout, N_, Fout, Fin, T_);
  launch_gemm(s, xin, Fin, (long long)N_ * Fin, Wq, Fout, (long long)Fin * Fout,
              bq, Fout, qbuf, Fout, (long long)N_ * Fout, N_, Fout, Fin, T_);
  launch_gemm(s, xin, Fin, (long long)N_ * Fin, Wv, Fout, (long long)Fin * Fout,
              bv, Fout, vbuf, Fout, (long long)N_ * Fout, N_, Fout, Fin, T_);

  fillu(s, aggbuf, 0u, (size_t)T_ * N_ * Fout);

  const float inv_sqrt_d = 1.0f / sqrtf((float)D);
  const int tot = E_ * H_;

  for (int e = 0; e < 4; ++e) {
    const int st = EDGE_ST[e], dt = EDGE_DT[e];
    const int* src = edges[e];
    const int* dst = edges[e] + E_;

    // per-head relation transforms (batch over heads, strided sub-matrices)
    launch_gemm(s, kbuf + (size_t)st * N_ * Fout, Fout, D,
                arel + (size_t)e * H_ * D * D, D, (long long)D * D,
                nullptr, 0, krel, Fout, D, N_, D, D, H_);
    launch_gemm(s, vbuf + (size_t)st * N_ * Fout, Fout, D,
                mrel + (size_t)e * H_ * D * D, D, (long long)D * D,
                nullptr, 0, vrel, Fout, D, N_, D, D, H_);

    fillu(s, segmax, 0xFF800000u, (size_t)N_ * H_);   // -inf bit pattern
    fillu(s, segsum, 0u, (size_t)N_ * H_);

    TCMHGT_edge_logits<<<(tot + 255) / 256, 256, 0, s>>>(
        qbuf + (size_t)dt * N_ * Fout, krel, src, dst, prel + e * H_,
        alpha, segmax, E_, H_, D, Fout, inv_sqrt_d);
    TCMHGT_edge_exp<<<(tot + 255) / 256, 256, 0, s>>>(
        alpha, segmax, segsum, dst, E_, H_);
    TCMHGT_edge_agg<<<E_, 256, 0, s>>>(
        alpha, segsum, vrel, src, dst,
        aggbuf + (size_t)dt * N_ * Fout, H_, D, Fout);
  }

  TCMHGT_gelu<<<2048, 256, 0, s>>>(aggbuf, gbuf, (size_t)T_ * N_ * Fout);

  launch_gemm(s, gbuf, Fout, (long long)N_ * Fout, Wa, Fout, (long long)Fout * Fout,
              ba, Fout, hbuf, Fout, (long long)N_ * Fout, N_, Fout, Fout, T_);
}

// ---------------------------------------------------------------------------
// Entry point
// ---------------------------------------------------------------------------
extern "C" void kernel_launch(void* const* d_in, const int* in_sizes, int n_in,
                              void* d_out, int out_size, void* d_ws, size_t ws_size,
                              hipStream_t stream)
{
  (void)in_sizes; (void)n_in; (void)out_size; (void)ws_size;

  const float* x    = (const float*)d_in[0];
  const int* edges[4] = {(const int*)d_in[1], (const int*)d_in[2],
                         (const int*)d_in[3], (const int*)d_in[4]};
  const float* Wk1  = (const float*)d_in[5];
  const float* bk1  = (const float*)d_in[6];
  const float* Wq1  = (const float*)d_in[7];
  const float* bq1  = (const float*)d_in[8];
  const float* Wv1  = (const float*)d_in[9];
  const float* bv1  = (const float*)d_in[10];
  const float* Wa1  = (const float*)d_in[11];
  const float* ba1  = (const float*)d_in[12];
  const float* arel1= (const float*)d_in[13];
  const float* mrel1= (const float*)d_in[14];
  const float* prel1= (const float*)d_in[15];
  const float* Wk2  = (const float*)d_in[16];
  const float* bk2  = (const float*)d_in[17];
  const float* Wq2  = (const float*)d_in[18];
  const float* bq2  = (const float*)d_in[19];
  const float* Wv2  = (const float*)d_in[20];
  const float* bv2  = (const float*)d_in[21];
  const float* Wa2  = (const float*)d_in[22];
  const float* ba2  = (const float*)d_in[23];
  const float* arel2= (const float*)d_in[24];
  const float* mrel2= (const float*)d_in[25];
  const float* prel2= (const float*)d_in[26];
  const float* Wlin = (const float*)d_in[27];
  const float* blin = (const float*)d_in[28];

  // ----- workspace layout (floats) ----------------------------------------
  const size_t NT = (size_t)T_ * N_;           // 20000
  const size_t SK = NT * 1024;                 // one [T,N,1024] buffer
  float* ws     = (float*)d_ws;
  float* kbuf   = ws;                          // SK
  float* qbuf   = ws + SK;                     // SK  (becomes h1)
  float* vbuf   = ws + 2 * SK;                 // SK
  float* aggbuf = ws + 3 * SK;                 // SK
  float* krel   = ws + 4 * SK;                 // N*1024
  float* vrel   = krel + (size_t)N_ * 1024;    // N*1024
  float* alpha  = vrel + (size_t)N_ * 1024;    // E*H
  float* segmax = alpha + (size_t)E_ * H_;     // N*H
  float* segsum = segmax + (size_t)N_ * H_;    // N*H
  // total: ~103.0M floats (~412 MB)

  // ----- layer 1: Fin=2048, Fout=1024, D=128 ------------------------------
  // g1 overwrites kbuf (k no longer needed after krel GEMMs),
  // h1 goes to qbuf (q no longer needed after logits kernels).
  run_layer(stream, x, FIN_, 2 * HID_, 128,
            Wk1, bk1, Wq1, bq1, Wv1, bv1, Wa1, ba1, arel1, mrel1, prel1,
            edges, kbuf, qbuf, vbuf, aggbuf,
            /*gbuf=*/kbuf, /*hbuf=*/qbuf,
            krel, vrel, alpha, segmax, segsum);
  const float* h1 = qbuf;

  // ----- layer 2: Fin=1024, Fout=512, D=64 --------------------------------
  const size_t SK2 = NT * 512;
  float* k2   = kbuf;
  float* q2   = kbuf + SK2;
  float* v2   = vbuf;
  float* agg2 = vbuf + SK2;
  float* g2   = aggbuf;
  float* h2   = aggbuf + SK2;
  run_layer(stream, h1, 2 * HID_, HID_, 64,
            Wk2, bk2, Wq2, bq2, Wv2, bv2, Wa2, ba2, arel2, mrel2, prel2,
            edges, k2, q2, v2, agg2,
            /*gbuf=*/g2, /*hbuf=*/h2,
            krel, vrel, alpha, segmax, segsum);

  // ----- final linear + softmax -------------------------------------------
  TCMHGT_head<<<((int)NT + 255) / 256, 256, 0, stream>>>(
      h2, Wlin, blin, (float*)d_out, (int)NT, HID_);
}
